// CaptionModel_71485435674989
// MI455X (gfx1250) — matrix operand order repaired and checked by reading, not compiled, not run on hardware
//
#include <hip/hip_runtime.h>
#include <math.h>

// ---------------------------------------------------------------------------
// CDNA5 (gfx1250) captioning model: ResNet-18 (implicit-GEMM bf16 WMMA convs,
// pre-packed bf16 weights, double-buffered LDS pipeline) -> degenerate LSTM
// scan fused into one persistent workgroup -> WMMA projection GEMM.
// ---------------------------------------------------------------------------

typedef __bf16 v16bf __attribute__((ext_vector_type(16)));
typedef __bf16 v8bf  __attribute__((ext_vector_type(8)));
typedef float  v8f   __attribute__((ext_vector_type(8)));

union F16x16 { v16bf v; v8bf h[2]; __bf16 e[16]; };

#define BN_EPS 1e-5f

// ===========================================================================
// Pack fp32 OIHW conv weights -> bf16 [cout][Kp] (K zero-padded to 32).
// Makes the conv A-fragment two contiguous 16B vector loads.
// ===========================================================================
__global__ void pack_conv_w(const float* __restrict__ w, __bf16* __restrict__ wp,
                            int K, int Kp, int total)
{
  int idx = blockIdx.x * 256 + threadIdx.x;
  if (idx >= total) return;
  int k = idx % Kp; int co = idx / Kp;
  wp[idx] = (__bf16)((k < K) ? w[(size_t)co * K + k] : 0.f);
}

// ===========================================================================
// Conv + BN (+residual) (+ReLU), implicit GEMM with bf16 WMMA.
// M = Cout (A = packed bf16 weights), N = B*OH*OW (im2col B-tile in LDS),
// K = Cin*KH*KW.  NW waves share one 64-pixel x 32-K LDS tile; each wave owns
// a 16-cout slice with 4 accumulators.  Double-buffered LDS: the next chunk's
// im2col gather is issued to VGPRs *before* the WMMAs on the current chunk,
// so global-load latency hides behind matrix math; one barrier per chunk.
// ===========================================================================
template <int KH, int KW, bool RAGGED, int NW>
__global__ __launch_bounds__(NW * 32) void conv_bn_wmma(
    const float* __restrict__ in, const __bf16* __restrict__ wp,
    const float* __restrict__ bng, const float* __restrict__ bnb,
    const float* __restrict__ bnm, const float* __restrict__ bnv,
    const float* __restrict__ res, float* __restrict__ out,
    int N, int Cin, int H, int W, int Cout,
    int stride, int pad, int OH, int OW, int relu)
{
  constexpr int KHW = KH * KW;
  constexpr int EPT = 2048 / (NW * 32);   // gather elements per thread
  constexpr int TPR = 32 / EPT;           // threads per LDS row
  const int Npix = N * OH * OW;
  const int K    = Cin * KHW;
  const int Kp   = RAGGED ? ((K + 31) & ~31) : K;
  const int ohw  = OH * OW;

  const int tid  = threadIdx.x;
  const int lane = tid & 31;
  const int wv   = tid >> 5;
  const int hf   = lane >> 4;
  const int ml   = lane & 15;
  const int nBase    = blockIdx.x * 64;
  const int coutBase = blockIdx.y * (NW * 16) + wv * 16;

  __shared__ alignas(32) __bf16 sB[2][64][32];   // double-buffered [n][k] tile

  v8f acc[4] = {};

  const int n_off = tid / TPR;
  const int k0    = (tid % TPR) * EPT;
  const int pix   = nBase + n_off;
  const bool pix_ok = pix < Npix;
  int gb = 0, ihb = 0, iwb = 0;
  if (pix_ok) {
    gb = pix / ohw; int rp = pix % ohw;
    ihb = (rp / OW) * stride - pad;
    iwb = (rp % OW) * stride - pad;
  }
  const float* inb = in + (size_t)gb * Cin * H * W;
  const __bf16* wr = wp + (size_t)(coutBase + ml) * Kp;

  // gather one 32-K chunk of the im2col tile into registers (no divisions:
  // compile-time divisor once, then incremental (cin,kr) walk)
  auto gather = [&](int kb, float (&nv)[EPT]) {
    int kk0 = kb + k0;
    int cin = kk0 / KHW;
    int kr  = kk0 - cin * KHW;
    #pragma unroll
    for (int e = 0; e < EPT; e++) {
      float val = 0.f;
      if (pix_ok && (!RAGGED || (kk0 + e) < K)) {
        int kh  = kr / KW;
        int kw2 = kr - kh * KW;
        int ih  = ihb + kh;
        int iw  = iwb + kw2;
        if ((KHW == 1) ||
            ((unsigned)ih < (unsigned)H && (unsigned)iw < (unsigned)W))
          val = inb[((size_t)cin * H + ih) * W + iw];
      }
      nv[e] = val;
      if (++kr == KHW) { kr = 0; cin++; }
    }
  };

  // prologue: stage chunk 0
  {
    float nv[EPT];
    gather(0, nv);
    #pragma unroll
    for (int e = 0; e < EPT; e++) sB[0][n_off][k0 + e] = (__bf16)nv[e];
  }
  __syncthreads();

  int buf = 0;
  for (int kb = 0; kb < Kp; kb += 32, buf ^= 1) {
    const bool has_next = (kb + 32) < Kp;
    float nv[EPT];
    if (has_next) gather(kb + 32, nv);      // loads issue here...

    // ...and WMMAs on the current chunk run while they are in flight
    F16x16 a;
    a.v = *(const v16bf*)&wr[kb + hf * 16];

    #pragma unroll
    for (int t = 0; t < 4; t++) {
      F16x16 bf;
      bf.v = *(const v16bf*)&sB[buf][t * 16 + ml][hf * 16];  // two ds_load_b128
      acc[t] = __builtin_amdgcn_wmma_f32_16x16x32_bf16(
          false, a.v, false, bf.v, (short)0, acc[t], false, false);
    }

    if (has_next) {
      #pragma unroll
      for (int e = 0; e < EPT; e++) sB[buf ^ 1][n_off][k0 + e] = (__bf16)nv[e];
    }
    __syncthreads();
  }

  // epilogue: fold BN, optional residual, optional ReLU
  float scale[8], shift[8];
  #pragma unroll
  for (int r = 0; r < 8; r++) {
    int co = coutBase + r + 8 * hf;
    float sc = bng[co] * rsqrtf(bnv[co] + BN_EPS);
    scale[r] = sc;
    shift[r] = bnb[co] - bnm[co] * sc;
  }
  for (int t = 0; t < 4; t++) {
    int p = nBase + t * 16 + ml;
    if (p >= Npix) continue;
    int b = p / ohw; int rp = p % ohw; int oh = rp / OW; int ow = rp % OW;
    #pragma unroll
    for (int r = 0; r < 8; r++) {
      int co = coutBase + r + 8 * hf;
      float vv = acc[t][r] * scale[r] + shift[r];
      size_t oidx = (((size_t)b * Cout + co) * OH + oh) * OW + ow;
      if (res)  vv += res[oidx];
      if (relu) vv = fmaxf(vv, 0.f);
      out[oidx] = vv;
    }
  }
}

// ===========================================================================
// 3x3/2 maxpool (pad 1), 112->56
// ===========================================================================
__global__ void maxpool_k(const float* __restrict__ in, float* __restrict__ out)
{
  int idx = blockIdx.x * 256 + threadIdx.x;
  const int total = 32 * 64 * 56 * 56;
  if (idx >= total) return;
  int ow = idx % 56; int t = idx / 56; int oh = t % 56; t /= 56;
  int c = t % 64; int n = t / 64;
  const float* base = in + (((size_t)n * 64 + c) * 112) * 112;
  float mx = -3.4e38f;
  for (int kh = 0; kh < 3; kh++) {
    int ih = oh * 2 - 1 + kh; if (ih < 0 || ih >= 112) continue;
    for (int kw = 0; kw < 3; kw++) {
      int iw = ow * 2 - 1 + kw; if (iw < 0 || iw >= 112) continue;
      mx = fmaxf(mx, base[ih * 112 + iw]);
    }
  }
  out[idx] = mx;
}

// global average pool over 7x7 -> feat[32][512]
__global__ void avgpool_k(const float* __restrict__ in, float* __restrict__ out)
{
  int idx = blockIdx.x * 256 + threadIdx.x;
  if (idx >= 32 * 512) return;
  const float* p = in + (size_t)idx * 49;
  float s = 0.f;
  for (int i = 0; i < 49; i++) s += p[i];
  out[idx] = s * (1.f / 49.f);
}

// ===========================================================================
// hgates = feat @ Wh[l].T + bh[l] + bi[l]  (h0==feat, c0==0 => computed once)
// grid (2048/16, 32/16, 2 layers), one wave per 16x16 tile
// ===========================================================================
__global__ __launch_bounds__(32) void hgates_wmma(
    const float* __restrict__ feat, const float* __restrict__ Wh,
    const float* __restrict__ bh, const float* __restrict__ bi,
    float* __restrict__ hg)
{
  const int l = blockIdx.z;
  const int nBase = blockIdx.x * 16;
  const int mBase = blockIdx.y * 16;
  const int lane = threadIdx.x;
  const int hf = lane >> 4, ml = lane & 15;
  const float* Wl = Wh + (size_t)l * 2048 * 512;
  v8f acc = {};
  for (int kb = 0; kb < 512; kb += 32) {
    F16x16 a, b;
    const float* fr = feat + (size_t)(mBase + ml) * 512;
    #pragma unroll
    for (int e = 0; e < 8; e++) {
      a.e[e]     = (__bf16)fr[kb + hf * 16 + e];
      a.e[e + 8] = (__bf16)fr[kb + hf * 16 + 8 + e];
    }
    const float* wr = Wl + (size_t)(nBase + ml) * 512 + kb + hf * 16;
    #pragma unroll
    for (int e = 0; e < 16; e++) b.e[e] = (__bf16)wr[e];
    acc = __builtin_amdgcn_wmma_f32_16x16x32_bf16(
        false, a.v, false, b.v, (short)0, acc, false, false);
  }
  int n = nBase + ml;
  float add = bh[l * 2048 + n] + bi[l * 2048 + n];
  #pragma unroll
  for (int r = 0; r < 8; r++) {
    int b_ = mBase + r + 8 * hf;
    hg[((size_t)l * 32 + b_) * 2048 + n] = acc[r] + add;
  }
}

// Repack Wi -> bf16, keeping only live gates {i,g,o} (forget gate dead: c0=0).
// Wb layout: [l][qi][n(512)][k(512)] bf16, fragment-contiguous in k.
__global__ void pack_wi(const float* __restrict__ Wi, __bf16* __restrict__ Wb)
{
  int idx = blockIdx.x * 256 + threadIdx.x;
  const int total = 2 * 3 * 512 * 512;
  if (idx >= total) return;
  int k = idx % 512; int t = idx / 512;
  int n = t % 512; t /= 512;
  int qi = t % 3; int l = t / 3;
  int q = (qi == 0) ? 0 : (qi == 1) ? 2 : 3;   // gate order i,f,g,o -> keep i,g,o
  Wb[idx] = (__bf16)Wi[((size_t)l * 2048 + q * 512 + n) * 512 + k];
}

// ===========================================================================
// Persistent LSTM scan: ONE workgroup (512 thr = 16 waves) loops 200 steps x
// 2 layers.  x kept in LDS as bf16 (double buffered), gate weights streamed
// from L2 in bf16 (with global_prefetch on the stream), layer sync via
// __syncthreads.  Eliminates 400 dependent kernel launches.
// ===========================================================================
__global__ __launch_bounds__(512) void lstm_persistent(
    const __bf16* __restrict__ Wb,   // [2][3][512][512]
    const float*  __restrict__ hg,   // [2][32][2048] (bh+bi folded)
    const float*  __restrict__ embed,
    float* __restrict__ outs)        // [200][32][512]
{
  __shared__ alignas(32) __bf16 sX[32 * 512];
  __shared__ alignas(32) __bf16 sY[32 * 512];
  const int tid  = threadIdx.x;
  const int wv   = tid >> 5;
  const int lane = tid & 31;
  const int hf = lane >> 4, ml = lane & 15;

  for (int i = tid; i < 32 * 512; i += 512)
    sX[i] = (__bf16)embed[512 + (i & 511)];     // SOS == 1, broadcast over batch
  __syncthreads();

  for (int t = 0; t < 200; t++) {
    #pragma unroll 1
    for (int l = 0; l < 2; l++) {
      const __bf16* xs = (l == 0) ? sX : sY;
      __bf16*       xd = (l == 0) ? sY : sX;
      const __bf16* Wl  = Wb + (size_t)l * 3 * 512 * 512;
      const float*  hgl = hg + (size_t)l * 32 * 2048;
      #pragma unroll 1
      for (int hn = 0; hn < 2; hn++) {
        const int nBase = (wv + hn * 16) * 16;
        const int n = nBase + ml;
        v8f acc[2][3] = {};
        for (int kb = 0; kb < 512; kb += 32) {
          F16x16 a0, a1;
          a0.h[0] = *(const v8bf*)&xs[ml * 512 + kb + hf * 16];
          a0.h[1] = *(const v8bf*)&xs[ml * 512 + kb + hf * 16 + 8];
          a1.h[0] = *(const v8bf*)&xs[(16 + ml) * 512 + kb + hf * 16];
          a1.h[1] = *(const v8bf*)&xs[(16 + ml) * 512 + kb + hf * 16 + 8];
          #pragma unroll
          for (int q = 0; q < 3; q++) {
            const __bf16* wrow = &Wl[((size_t)q * 512 + n) * 512 + kb + hf * 16];
            if (kb + 64 < 512)
              __builtin_prefetch(wrow + 64, 0, 1);   // global_prefetch_b8
            F16x16 bfr;
            bfr.v = *(const v16bf*)wrow;
            acc[0][q] = __builtin_amdgcn_wmma_f32_16x16x32_bf16(
                false, a0.v, false, bfr.v, (short)0, acc[0][q], false, false);
            acc[1][q] = __builtin_amdgcn_wmma_f32_16x16x32_bf16(
                false, a1.v, false, bfr.v, (short)0, acc[1][q], false, false);
          }
        }
        #pragma unroll
        for (int mT = 0; mT < 2; mT++) {
          #pragma unroll
          for (int r = 0; r < 8; r++) {
            int b = mT * 16 + r + 8 * hf;
            const float* hb = hgl + (size_t)b * 2048;
            float gi = acc[mT][0][r] + hb[n];            // gate i
            float gg = acc[mT][1][r] + hb[1024 + n];     // gate g
            float go = acc[mT][2][r] + hb[1536 + n];     // gate o
            float iv = 1.f / (1.f + __expf(-gi));
            float gv = tanhf(gg);
            float ov = 1.f / (1.f + __expf(-go));
            float xv = ov * tanhf(iv * gv);              // c0==0 => f-gate dead
            xd[b * 512 + n] = (__bf16)xv;
            if (l == 1) outs[(size_t)t * 16384 + b * 512 + n] = xv;
          }
        }
      }
      __syncthreads();
    }
  }
}

// ===========================================================================
// logits[b][v][t] = outs[t][b][:] . proj_w[v][:] + proj_b[v]
// M = B*STEPS = 6400 (m = b*200+t), N = 128, K = 512.  grid (400, 8), 1 wave.
// ===========================================================================
__global__ __launch_bounds__(32) void proj_wmma(
    const float* __restrict__ outs, const float* __restrict__ pw,
    const float* __restrict__ pb, float* __restrict__ logits)
{
  const int mBase = blockIdx.x * 16;
  const int nBase = blockIdx.y * 16;
  const int lane = threadIdx.x;
  const int hf = lane >> 4, ml = lane & 15;
  v8f acc = {};
  for (int kb = 0; kb < 512; kb += 32) {
    F16x16 a, b;
    {
      int m = mBase + ml;
      int bb = m / 200, tt = m % 200;
      const float* ar = outs + ((size_t)tt * 32 + bb) * 512;
      #pragma unroll
      for (int e = 0; e < 8; e++) {
        a.e[e]     = (__bf16)ar[kb + hf * 16 + e];
        a.e[e + 8] = (__bf16)ar[kb + hf * 16 + 8 + e];
      }
    }
    const float* wr = pw + (size_t)(nBase + ml) * 512 + kb + hf * 16;
    #pragma unroll
    for (int e = 0; e < 16; e++) b.e[e] = (__bf16)wr[e];
    acc = __builtin_amdgcn_wmma_f32_16x16x32_bf16(
        false, a.v, false, b.v, (short)0, acc, false, false);
  }
  int v = nBase + ml;
  float bias = pb[v];
  #pragma unroll
  for (int r = 0; r < 8; r++) {
    int m = mBase + r + 8 * hf;
    int bb = m / 200, tt = m % 200;
    logits[((size_t)bb * 128 + v) * 200 + tt] = acc[r] + bias;
  }
}

// ===========================================================================
// Host orchestration
// ===========================================================================
struct BNp { const float *g, *b, *m, *v; };

static BNp getbn(void* const* din, int& i)
{
  BNp r{ (const float*)din[i], (const float*)din[i + 1],
         (const float*)din[i + 2], (const float*)din[i + 3] };
  i += 4;
  return r;
}

struct WPack { hipStream_t s; __bf16* pool; size_t off; };

static __bf16* pack_w(WPack& c, const float* w, int Cout, int K)
{
  int Kp = (K + 31) & ~31;
  __bf16* dst = c.pool + c.off;
  c.off += (size_t)Cout * Kp;
  int total = Cout * Kp;
  pack_conv_w<<<(total + 255) / 256, 256, 0, c.s>>>(w, dst, K, Kp, total);
  return dst;
}

static void launch_conv(hipStream_t s, const float* in, const __bf16* wp, BNp bn,
                        const float* res, float* out,
                        int N, int Cin, int H, int W, int Cout, int KH, int KW,
                        int stride, int pad, int relu)
{
  int OH = (H + 2 * pad - KH) / stride + 1;
  int OW = (W + 2 * pad - KW) / stride + 1;
  int Npix = N * OH * OW;
  if (KH == 7) {                        // stem: K=147 ragged, Cout=64
    dim3 grid((Npix + 63) / 64, Cout / 64);
    conv_bn_wmma<7, 7, true, 4><<<grid, 128, 0, s>>>(
        in, wp, bn.g, bn.b, bn.m, bn.v, res, out,
        N, Cin, H, W, Cout, stride, pad, OH, OW, relu);
  } else if (KH == 3) {
    if (Cout >= 128) {
      dim3 grid((Npix + 63) / 64, Cout / 128);
      conv_bn_wmma<3, 3, false, 8><<<grid, 256, 0, s>>>(
          in, wp, bn.g, bn.b, bn.m, bn.v, res, out,
          N, Cin, H, W, Cout, stride, pad, OH, OW, relu);
    } else {
      dim3 grid((Npix + 63) / 64, Cout / 64);
      conv_bn_wmma<3, 3, false, 4><<<grid, 128, 0, s>>>(
          in, wp, bn.g, bn.b, bn.m, bn.v, res, out,
          N, Cin, H, W, Cout, stride, pad, OH, OW, relu);
    }
  } else {                              // 1x1 downsamples, Cout>=128
    dim3 grid((Npix + 63) / 64, Cout / 128);
    conv_bn_wmma<1, 1, false, 8><<<grid, 256, 0, s>>>(
        in, wp, bn.g, bn.b, bn.m, bn.v, res, out,
        N, Cin, H, W, Cout, stride, pad, OH, OW, relu);
  }
}

extern "C" void kernel_launch(void* const* d_in, const int* in_sizes, int n_in,
                              void* d_out, int out_size, void* d_ws, size_t ws_size,
                              hipStream_t stream)
{
  (void)in_sizes; (void)n_in; (void)out_size; (void)ws_size;

  // ---- unpack inputs (setup_inputs dict insertion order, depth-first) ----
  int i = 0;
  const float* img    = (const float*)d_in[i++];
  const float* stem_w = (const float*)d_in[i++];
  BNp stem_bn = getbn(d_in, i);

  struct Blk { const float* w1; BNp bn1; const float* w2; BNp bn2;
               const float* dw; BNp dbn; bool down; };
  Blk blk[4][2];
  const int chans[4] = {64, 128, 256, 512};
  int inc = 64;
  for (int s = 0; s < 4; s++) {
    int c = chans[s];
    for (int b = 0; b < 2; b++) {
      int st = (s > 0 && b == 0) ? 2 : 1;
      Blk& B = blk[s][b];
      B.w1 = (const float*)d_in[i++]; B.bn1 = getbn(d_in, i);
      B.w2 = (const float*)d_in[i++]; B.bn2 = getbn(d_in, i);
      B.down = (st != 1 || inc != c);
      B.dw = nullptr;
      if (B.down) { B.dw = (const float*)d_in[i++]; B.dbn = getbn(d_in, i); }
      inc = c;
    }
  }
  const float* Wi     = (const float*)d_in[i++];
  const float* Wh     = (const float*)d_in[i++];
  const float* bi     = (const float*)d_in[i++];
  const float* bh     = (const float*)d_in[i++];
  const float* proj_w = (const float*)d_in[i++];
  const float* proj_b = (const float*)d_in[i++];
  const float* embed  = (const float*)d_in[i++];

  // ---- workspace carve (floats, 64-elem aligned); ~195 MB total ----
  float* ws = (float*)d_ws;
  size_t off = 0;
  auto alloc = [&](size_t n) { float* p = ws + off; off += (n + 63) & ~(size_t)63; return p; };
  float* buf0  = alloc((size_t)32 * 64 * 112 * 112);   // stem out (largest)
  float* buf1  = alloc((size_t)32 * 64 * 56 * 56);
  float* buf2  = alloc((size_t)32 * 64 * 56 * 56);
  float* feat  = alloc(32 * 512);
  float* hg2   = alloc(2 * 32 * 2048);
  float* outsb = alloc((size_t)200 * 32 * 512);
  __bf16* Wb   = (__bf16*)alloc((size_t)2 * 3 * 512 * 512 / 2);
  __bf16* wpool = (__bf16*)alloc(5600000);             // 11.2M bf16 conv weights

  // ---- pack all conv weights to bf16 (in-order on stream) ----
  WPack wc{stream, wpool, 0};
  __bf16* stem_wp = pack_w(wc, stem_w, 64, 3 * 49);
  __bf16* w1p[4][2]; __bf16* w2p[4][2]; __bf16* dwp[4][2];
  inc = 64;
  for (int s = 0; s < 4; s++) {
    int c = chans[s];
    for (int b = 0; b < 2; b++) {
      Blk& B = blk[s][b];
      w1p[s][b] = pack_w(wc, B.w1, c, inc * 9);
      w2p[s][b] = pack_w(wc, B.w2, c, c * 9);
      dwp[s][b] = B.down ? pack_w(wc, B.dw, c, inc) : nullptr;
      inc = c;
    }
  }

  // ---- ResNet-18 ----
  launch_conv(stream, img, stem_wp, stem_bn, nullptr, buf0,
              32, 3, 224, 224, 64, 7, 7, 2, 3, 1);
  {
    int total = 32 * 64 * 56 * 56;
    maxpool_k<<<(total + 255) / 256, 256, 0, stream>>>(buf0, buf1);
  }

  float* bufs[3] = { buf0, buf1, buf2 };
  int cur = 1;                 // pool output lives in buf1
  int Cin = 64, Hh = 56, Ww = 56;
  for (int s = 0; s < 4; s++) {
    int c = chans[s];
    for (int b = 0; b < 2; b++) {
      int st = (s > 0 && b == 0) ? 2 : 1;
      Blk& B = blk[s][b];
      int A = (cur + 1) % 3, Bn = (cur + 2) % 3;
      int OHW = (Hh + 2 - 3) / st + 1;
      launch_conv(stream, bufs[cur], w1p[s][b], B.bn1, nullptr, bufs[A],
                  32, Cin, Hh, Ww, c, 3, 3, st, 1, 1);
      if (B.down) {
        launch_conv(stream, bufs[cur], dwp[s][b], B.dbn, nullptr, bufs[Bn],
                    32, Cin, Hh, Ww, c, 1, 1, st, 0, 0);
        launch_conv(stream, bufs[A], w2p[s][b], B.bn2, bufs[Bn], bufs[cur],
                    32, c, OHW, OHW, c, 3, 3, 1, 1, 1);
        // output stays in bufs[cur]
      } else {
        launch_conv(stream, bufs[A], w2p[s][b], B.bn2, bufs[cur], bufs[Bn],
                    32, c, OHW, OHW, c, 3, 3, 1, 1, 1);
        cur = Bn;
      }
      Cin = c; Hh = OHW; Ww = OHW;
    }
  }
  avgpool_k<<<(32 * 512 + 255) / 256, 256, 0, stream>>>(bufs[cur], feat);

  // ---- LSTM prep + persistent scan ----
  {
    int total = 2 * 3 * 512 * 512;
    pack_wi<<<(total + 255) / 256, 256, 0, stream>>>(Wi, Wb);
  }
  hgates_wmma<<<dim3(128, 2, 2), 32, 0, stream>>>(feat, Wh, bh, bi, hg2);
  lstm_persistent<<<1, 512, 0, stream>>>(Wb, hg2, embed, outsb);

  // ---- projection -> [B,V,STEPS] ----
  proj_wmma<<<dim3(400, 8), 32, 0, stream>>>(outsb, proj_w, proj_b, (float*)d_out);
}
